// Model_19061064860127
// MI455X (gfx1250) — compile-verified, matrix operand order checked by reading, not run on hardware
//
#include <hip/hip_runtime.h>
#include <hip/hip_bf16.h>

#define NA 100000
#define NT 50000
#define F_IN 32
#define NE 1000000
#define NEL 500000
#define BN_EPS 1e-5f

typedef __attribute__((ext_vector_type(16))) _Float16 v16h;
typedef __attribute__((ext_vector_type(8)))  float    v8f;
typedef __attribute__((ext_vector_type(4)))  unsigned u32x4;
typedef __attribute__((ext_vector_type(8)))  unsigned u32x8;
typedef _Float16 h16;

union V16U { uint4 q[2]; v16h v; };

// ---------------------------------------------------------------------------
// CDNA5 data movers
// ---------------------------------------------------------------------------
// 1-D TDM copy: nwords dwords from gsrc -> LDS offset lds_off. D# per
// CDNA5 ISA. Issued once per wave (EXEC ignored).
__device__ __forceinline__ void tdm_load_1d(unsigned lds_off, const void* gsrc,
                                            unsigned nwords) {
    unsigned long long ga = (unsigned long long)gsrc;
    u32x4 g0;
    g0[0] = 1u;                                           // count=1 user desc
    g0[1] = lds_off;                                      // lds_addr
    g0[2] = (unsigned)ga;                                 // global_addr[31:0]
    g0[3] = (unsigned)((ga >> 32) & 0x01FFFFFFu) | (2u << 30);  // [56:32]|type=2
    u32x8 g1;
    g1[0] = (2u << 16);                                   // data_size = 4B
    g1[1] = (nwords & 0xFFFFu) << 16;                     // tensor_dim0 lo
    g1[2] = (nwords >> 16) | (1u << 16);                  // dim0 hi | tensor_dim1=1
    g1[3] = (nwords & 0xFFFFu) << 16;                     // tile_dim0
    g1[4] = 1u;                                           // tile_dim1=1
    g1[5] = nwords;                                       // tensor_dim0_stride lo
    g1[6] = 0u;
    g1[7] = 0u;
    asm volatile("tensor_load_to_lds %0, %1" :: "s"(g0), "s"(g1) : "memory");
}

__device__ __forceinline__ void async_b128(unsigned lds_off, const void* g) {
    asm volatile("global_load_async_to_lds_b128 %0, %1, off"
                 :: "v"(lds_off), "v"((unsigned long long)g) : "memory");
}
__device__ __forceinline__ void wait_async0() {
    asm volatile("s_wait_asynccnt 0x0" ::: "memory");
}
__device__ __forceinline__ unsigned lds_off_of(const void* p) {
    return (unsigned)(unsigned long long)p;   // low 32 bits = LDS offset
}

// swizzle helper: CDNA5 16-bit WMMA operand layout
// k2 (0..31) -> (g, j*2+r):  k2 = 16*(j>>2) + 8*g + 2*(j&3) + r
__device__ __forceinline__ void swz16(int k2, int& g, int& jr) {
    int quad = k2 >> 4, w = k2 & 15;
    g = w >> 3;
    int rem = w & 7;
    jr = (quad * 4 + (rem >> 1)) * 2 + (rem & 1);
}

// ---------------------------------------------------------------------------
// Tiled WMMA GEMM: C[M,64] = act(A[M,K] @ B[K,64] + bias)
// Block = 128 thr (4 waves) covering 64 rows x 64 cols: wave w owns rows
// [64b+16w, 64b+16w+16), sweeping all four 16-col tiles (4 accumulators).
// B panel (pre-swizzled) is TDM-loaded to LDS once per block; A slabs are
// swizzled into per-lane-contiguous LDS so each operand is two ds_load_b128.
// ---------------------------------------------------------------------------
template<int K, bool AF32, int ACT, bool BIAS, bool OUTF, bool OUTH>
__global__ __launch_bounds__(128)
void gemm_wmma_f16(const void* __restrict__ Aptr, const h16* __restrict__ Bop,
                   const float* __restrict__ bias,
                   float* __restrict__ C, h16* __restrict__ C16, int M)
{
    __shared__ __align__(32) h16 sB[K * 64];       // swizzled: [(ks*64+c)*2+g][16]
    __shared__ __align__(32) h16 sA[4 * 32 * 16];  // [wave][lane][16]
    const int tid  = threadIdx.x;
    const int wave = tid >> 5;
    const int lane = tid & 31;
    const int g    = lane >> 4;
    const int l15  = lane & 15;
    const int rowBase = blockIdx.x * 64;

    if (tid == 0) tdm_load_1d(lds_off_of(&sB[0]), Bop, (unsigned)(K * 32));

    v8f acc[4] = {};
#pragma unroll
    for (int kk = 0; kk < K; kk += 32) {
        __syncthreads();
        // stage + swizzle 64x32 A slab (16 elements/thread)
#pragma unroll
        for (int i = tid; i < 2048; i += 128) {
            int m = i >> 5, k2 = i & 31;
            int ga, jr; swz16(k2, ga, jr);
            int row = rowBase + m; if (row >= M) row = M - 1;  // clamp reads
            size_t gidx = (size_t)row * K + kk + k2;
            h16 v = AF32 ? (h16)(((const float*)Aptr)[gidx])
                         : ((const h16*)Aptr)[gidx];
            sA[(((m >> 4) * 32) + ga * 16 + (m & 15)) * 16 + jr] = v;
        }
        if (kk == 0) __builtin_amdgcn_s_wait_tensorcnt(0);
        __syncthreads();

        V16U a;
        const uint4* pa = (const uint4*)(sA + (wave * 32 + lane) * 16);
        a.q[0] = pa[0]; a.q[1] = pa[1];
#pragma unroll
        for (int ct = 0; ct < 4; ++ct) {
            V16U b;
            const uint4* pb = (const uint4*)(
                sB + (((kk >> 5) * 64 + ct * 16 + l15) * 2 + g) * 16);
            b.q[0] = pb[0]; b.q[1] = pb[1];
            acc[ct] = __builtin_amdgcn_wmma_f32_16x16x32_f16(
                false, a.v, false, b.v, (short)0, acc[ct], false, false);
        }
    }

#pragma unroll
    for (int ct = 0; ct < 4; ++ct) {
        const int col = ct * 16 + l15;
        const float bv = BIAS ? bias[col] : 0.0f;
#pragma unroll
        for (int i = 0; i < 8; ++i) {
            const int row = rowBase + wave * 16 + g * 8 + i;  // C/D: M = 8*g+i
            if (row < M) {
                float v = acc[ct][i] + bv;
                if (ACT == 1) v = fmaxf(v, 0.0f);
                size_t o = (size_t)row * 64 + col;
                if (OUTF) C[o]   = v;
                if (OUTH) C16[o] = (h16)v;
            }
        }
    }
}

// ---------------------------------------------------------------------------
// Weight conversion: f32 W[K][64] -> f16 pre-swizzled WMMA B-operand panel
// ---------------------------------------------------------------------------
__global__ void cvt_swz_B_k(const float* __restrict__ W, h16* __restrict__ Bop, int K) {
    int i = blockIdx.x * blockDim.x + threadIdx.x;
    if (i >= K * 64) return;
    int k = i >> 6, c = i & 63;
    int ks = k >> 5, k2 = k & 31;
    int g, jr; swz16(k2, g, jr);
    Bop[((ks * 64 + c) * 2 + g) * 16 + jr] = (h16)W[i];
}

// ---------------------------------------------------------------------------
// Elementwise / helper kernels
// ---------------------------------------------------------------------------
__global__ void fill_f32_k(float* __restrict__ p, float v, int n) {
    int i = blockIdx.x * blockDim.x + threadIdx.x;
    if (i < n) p[i] = v;
}

__global__ void gelu_cvt_k(const float* __restrict__ x, h16* __restrict__ y, int n) {
    int i = blockIdx.x * blockDim.x + threadIdx.x;
    if (i < n) {
        float v = x[i];
        y[i] = (h16)(0.5f * v * (1.0f + erff(v * 0.70710678118654752f)));
    }
}

__global__ void skip_mix_k(const float* __restrict__ a, const float* __restrict__ x,
                           const float* __restrict__ skip, float* __restrict__ out, int n) {
    int i = blockIdx.x * blockDim.x + threadIdx.x;
    if (i < n) {
        float s = 1.0f / (1.0f + __expf(-skip[0]));
        out[i] = s * a[i] + (1.0f - s) * x[i];
    }
}

__global__ __launch_bounds__(256)
void bn_stats_k(const float* __restrict__ x, int nrows,
                float* __restrict__ mu, float* __restrict__ var) {
    const int d = blockIdx.x;  // 0..63
    float s = 0.0f, s2 = 0.0f;
    for (int r = threadIdx.x; r < nrows; r += 256) {
        float v = x[(size_t)r * 64 + d];
        s += v; s2 += v * v;
    }
    __shared__ float sh[256], sh2[256];
    sh[threadIdx.x] = s; sh2[threadIdx.x] = s2;
    __syncthreads();
    for (int o = 128; o > 0; o >>= 1) {
        if (threadIdx.x < o) {
            sh[threadIdx.x]  += sh[threadIdx.x + o];
            sh2[threadIdx.x] += sh2[threadIdx.x + o];
        }
        __syncthreads();
    }
    if (threadIdx.x == 0) {
        float m = sh[0] / (float)nrows;
        mu[d]  = m;
        var[d] = sh2[0] / (float)nrows - m * m;  // biased variance
    }
}

__global__ void bn_apply_k(const float* __restrict__ x, const float* __restrict__ mu,
                           const float* __restrict__ var, const float* __restrict__ gamma,
                           const float* __restrict__ beta, float* __restrict__ y,
                           h16* __restrict__ y16, int n) {
    int i = blockIdx.x * blockDim.x + threadIdx.x;
    if (i < n) {
        int d = i & 63;
        float v = (x[i] - mu[d]) * rsqrtf(var[d] + BN_EPS) * gamma[d] + beta[d];
        y[i]   = v;
        y16[i] = (h16)v;
    }
}

// ---------------------------------------------------------------------------
// Edge kernels: attention logits + segment softmax + weighted scatter
// ---------------------------------------------------------------------------
__device__ __forceinline__ void atomicMaxF(float* addr, float v) {
    unsigned* ua = (unsigned*)addr;
    unsigned old = __hip_atomic_load(ua, __ATOMIC_RELAXED, __HIP_MEMORY_SCOPE_AGENT);
    while (__uint_as_float(old) < v) {
        unsigned assumed = old;
        old = atomicCAS(ua, assumed, __float_as_uint(v));
        if (old == assumed) break;
    }
}

__global__ void edge_alpha_k(const int* __restrict__ ei, int ne,
                             const float* __restrict__ q, const float* __restrict__ kt,
                             const float* __restrict__ p_rel,
                             float* __restrict__ alpha, float* __restrict__ mmax) {
    int e = blockIdx.x * blockDim.x + threadIdx.x;
    if (e >= ne) return;
    int r = ei[e], c = ei[ne + e];
    const float4* qa = (const float4*)(q  + (size_t)c * 64);
    const float4* ka = (const float4*)(kt + (size_t)r * 64);
    float s = 0.0f;
#pragma unroll
    for (int j = 0; j < 16; ++j) {
        float4 a = qa[j], b = ka[j];
        s += a.x * b.x + a.y * b.y + a.z * b.z + a.w * b.w;
    }
    s *= p_rel[0] * 0.125f;   // * p_rel / sqrt(D)
    alpha[e] = s;
    atomicMaxF(mmax + c, s);
}

__global__ void edge_exp_k(const int* __restrict__ ei, int ne,
                           float* __restrict__ alpha, const float* __restrict__ mmax,
                           float* __restrict__ den) {
    int e = blockIdx.x * blockDim.x + threadIdx.x;
    if (e >= ne) return;
    int c = ei[ne + e];
    float ex = __expf(alpha[e] - mmax[c]);
    alpha[e] = ex;
    atomicAdd(den + c, ex);
}

__global__ void edge_scatter_k(const int* __restrict__ ei, int ne,
                               const float* __restrict__ ex, const float* __restrict__ den,
                               const float* __restrict__ vt, float* __restrict__ agg) {
    int t = blockIdx.x * blockDim.x + threadIdx.x;
    int e = t >> 5, lane = t & 31;
    if (e >= ne) return;
    int r = ei[e], c = ei[ne + e];
    float w = ex[e] / den[c];
#pragma unroll
    for (int h = 0; h < 2; ++h) {
        int d = lane + 32 * h;
        atomicAdd(agg + (size_t)c * 64 + d, w * vt[(size_t)r * 64 + d]);
    }
}

// ---------------------------------------------------------------------------
// Fused decoder: 64 label edges/block (16 per wave). Gather zc=[z[r]||z[c]]
// (f32->f16, swizzled) into LDS; W1 panel (pre-swizzled) arrives via async
// b128 copies once per block; per wave: 4 k-steps x 4 col-tiles of WMMA for
// lin1 + bias + ReLU into LDS; then one thread per edge does the 64-dot lin2.
// ---------------------------------------------------------------------------
__global__ __launch_bounds__(128)
void decoder_k(const float* __restrict__ z, const int* __restrict__ eli,
               const h16* __restrict__ W1op, const float* __restrict__ b1,
               const float* __restrict__ W2, const float* __restrict__ b2,
               float* __restrict__ out) {
    __shared__ __align__(32) h16  sW[128 * 64];        // swizzled B panel (16KB)
    __shared__ __align__(32) h16  sA[16 * 32 * 16];    // [wave][ks][lane][16] (16KB)
    __shared__ float sH[64 * 64];                       // lin1 output (16KB)
    const int tid = threadIdx.x;
    const int e0  = blockIdx.x * 64;

    // async-copy the 16KB W1 panel (8 x b128 per thread)
    const unsigned ldsW = lds_off_of(&sW[0]);
#pragma unroll
    for (int it = 0; it < 8; ++it) {
        unsigned byteOff = (unsigned)(it * 128 + tid) * 16u;
        async_b128(ldsW + byteOff, (const char*)W1op + byteOff);
    }

    // gather + swizzle A: 64 edges x 128 features (64 elements/thread)
    for (int i = tid; i < 64 * 128; i += 128) {
        int er = i >> 7, d = i & 127;
        int e = e0 + er; if (e >= NEL) e = 0;           // clamp reads
        int node = (d < 64) ? eli[e] : eli[NEL + e];
        float v = z[(size_t)node * 64 + (d & 63)];
        int ks = d >> 5, k2 = d & 31;
        int ga, jr; swz16(k2, ga, jr);
        sA[((((er >> 4) * 4 + ks) * 32) + ga * 16 + (er & 15)) * 16 + jr] = (h16)v;
    }
    wait_async0();
    __syncthreads();

    const int wave = tid >> 5, lane = tid & 31;
    const int g = lane >> 4, l15 = lane & 15;
    v8f acc[4] = {};
#pragma unroll
    for (int ks = 0; ks < 4; ++ks) {
        V16U a;
        const uint4* pa = (const uint4*)(sA + ((wave * 4 + ks) * 32 + lane) * 16);
        a.q[0] = pa[0]; a.q[1] = pa[1];
#pragma unroll
        for (int ct = 0; ct < 4; ++ct) {
            V16U b;
            const uint4* pb = (const uint4*)(
                sW + ((ks * 64 + ct * 16 + l15) * 2 + g) * 16);
            b.q[0] = pb[0]; b.q[1] = pb[1];
            acc[ct] = __builtin_amdgcn_wmma_f32_16x16x32_f16(
                false, a.v, false, b.v, (short)0, acc[ct], false, false);
        }
    }
#pragma unroll
    for (int ct = 0; ct < 4; ++ct) {
        const int col = ct * 16 + l15;
        const float bv = b1[col];
#pragma unroll
        for (int i = 0; i < 8; ++i)
            sH[(wave * 16 + g * 8 + i) * 64 + col] = fmaxf(acc[ct][i] + bv, 0.0f);
    }
    __syncthreads();

    if (tid < 64) {
        int e = e0 + tid;
        if (e < NEL) {
            float s = b2[0];
#pragma unroll 8
            for (int j = 0; j < 64; ++j) s += sH[tid * 64 + j] * W2[j];
            out[e] = s;
        }
    }
}

// ---------------------------------------------------------------------------
// Host orchestration
// ---------------------------------------------------------------------------
extern "C" void kernel_launch(void* const* d_in, const int* in_sizes, int n_in,
                              void* d_out, int out_size, void* d_ws, size_t ws_size,
                              hipStream_t stream) {
    (void)in_sizes; (void)n_in; (void)out_size; (void)ws_size;
    int p = 0;
    auto NXT = [&]() { return (const float*)d_in[p++]; };

    const float* x_in[2];          // author, topic raw features
    x_in[0] = NXT(); x_in[1] = NXT();
    const float *linW[2], *linB[2];
    for (int t = 0; t < 2; ++t) { linW[t] = NXT(); linB[t] = NXT(); }

    struct LayerP {
        const float *kW[2], *kB[2], *qW[2], *qB[2], *vW[2], *vB[2], *aW[2], *aB[2];
        const float *skip[2];
        const float *aRel[2], *mRel[2], *pRel[2];   // [about, rev_about]
        const float *gamma, *beta;
    } L[2];
    for (int l = 0; l < 2; ++l) {
        for (int t = 0; t < 2; ++t) { L[l].kW[t] = NXT(); L[l].kB[t] = NXT(); }
        for (int t = 0; t < 2; ++t) { L[l].qW[t] = NXT(); L[l].qB[t] = NXT(); }
        for (int t = 0; t < 2; ++t) { L[l].vW[t] = NXT(); L[l].vB[t] = NXT(); }
        for (int t = 0; t < 2; ++t) { L[l].aW[t] = NXT(); L[l].aB[t] = NXT(); }
        for (int t = 0; t < 2; ++t) { L[l].skip[t] = NXT(); }
        for (int r = 0; r < 2; ++r) { L[l].aRel[r] = NXT(); L[l].mRel[r] = NXT(); L[l].pRel[r] = NXT(); }
        L[l].gamma = NXT(); L[l].beta = NXT();
    }
    const float *decW1 = NXT(), *decB1 = NXT(), *decW2 = NXT(), *decB2 = NXT();
    const int* eat = (const int*)d_in[p++];
    const int* eta = (const int*)d_in[p++];
    const int* eli = (const int*)d_in[p++];

    // ---- workspace carve-out --------------------------------------------
    char* wsb = (char*)d_ws;
    size_t off = 0;
    auto alloc = [&](size_t bytes) -> void* {
        void* q = wsb + off; off += (bytes + 255) & ~(size_t)255; return q;
    };
    const int N[2] = {NA, NT};
    float* xF[2];  h16* x16[2];  float* qF[2];  h16* k16[2]; h16* v16[2]; float* agg[2];
    for (int t = 0; t < 2; ++t) {
        xF[t]  = (float*)alloc((size_t)N[t] * 64 * 4);
        x16[t] = (h16*)  alloc((size_t)N[t] * 64 * 2);
        qF[t]  = (float*)alloc((size_t)N[t] * 64 * 4);
        k16[t] = (h16*)  alloc((size_t)N[t] * 64 * 2);
        v16[t] = (h16*)  alloc((size_t)N[t] * 64 * 2);
        agg[t] = (float*)alloc((size_t)N[t] * 64 * 4);
    }
    float* ktF   = (float*)alloc((size_t)NA * 64 * 4);   // reused as a-lin out (author)
    float* vtF   = (float*)alloc((size_t)NA * 64 * 4);   // reused as a-lin out (topic)
    float* alpha = (float*)alloc((size_t)NE * 4);
    float* mbuf  = (float*)alloc((size_t)NA * 4);
    float* den   = (float*)alloc((size_t)NA * 4);
    h16*  gelu16 = (h16*)  alloc((size_t)NA * 64 * 2);
    float* mu    = (float*)alloc(64 * 4);
    float* var   = (float*)alloc(64 * 4);

    auto allocH = [&](size_t n) { return (h16*)alloc(n * 2); };
    h16* linW16[2]; for (int t = 0; t < 2; ++t) linW16[t] = allocH(F_IN * 64);
    h16 *kW16[2][2], *qW16[2][2], *vW16[2][2], *aW16[2][2], *aRel16[2][2], *mRel16[2][2];
    for (int l = 0; l < 2; ++l)
        for (int t = 0; t < 2; ++t) {
            kW16[l][t] = allocH(64 * 64); qW16[l][t] = allocH(64 * 64);
            vW16[l][t] = allocH(64 * 64); aW16[l][t] = allocH(64 * 64);
            aRel16[l][t] = allocH(64 * 64); mRel16[l][t] = allocH(64 * 64);
        }
    h16* decW116 = allocH(128 * 64);

    auto cdiv = [](long a, long b) { return (unsigned)((a + b - 1) / b); };
    auto SWZ = [&](const float* s, h16* d, int K) {
        cvt_swz_B_k<<<cdiv((long)K * 64, 256), 256, 0, stream>>>(s, d, K);
    };

    // ---- weight conversion (f32 -> pre-swizzled f16 B panels) -----------
    for (int t = 0; t < 2; ++t) SWZ(linW[t], linW16[t], F_IN);
    for (int l = 0; l < 2; ++l)
        for (int t = 0; t < 2; ++t) {
            SWZ(L[l].kW[t], kW16[l][t], 64);
            SWZ(L[l].qW[t], qW16[l][t], 64);
            SWZ(L[l].vW[t], vW16[l][t], 64);
            SWZ(L[l].aW[t], aW16[l][t], 64);
            SWZ(L[l].aRel[t], aRel16[l][t], 64);
            SWZ(L[l].mRel[t], mRel16[l][t], 64);
        }
    SWZ(decW1, decW116, 128);

    // ---- input projection: x = relu(lin_in(x_raw)), keep f32 + f16 -------
    for (int t = 0; t < 2; ++t)
        gemm_wmma_f16<F_IN, true, 1, true, true, true><<<cdiv(N[t], 64), 128, 0, stream>>>(
            x_in[t], linW16[t], linB[t], xF[t], x16[t], N[t]);

    // ---- HGT layers ------------------------------------------------------
    for (int l = 0; l < 2; ++l) {
        for (int t = 0; t < 2; ++t) {
            gemm_wmma_f16<64, false, 0, true, true, false><<<cdiv(N[t], 64), 128, 0, stream>>>(
                x16[t], qW16[l][t], L[l].qB[t], qF[t], nullptr, N[t]);
            gemm_wmma_f16<64, false, 0, true, false, true><<<cdiv(N[t], 64), 128, 0, stream>>>(
                x16[t], kW16[l][t], L[l].kB[t], nullptr, k16[t], N[t]);
            gemm_wmma_f16<64, false, 0, true, false, true><<<cdiv(N[t], 64), 128, 0, stream>>>(
                x16[t], vW16[l][t], L[l].vB[t], nullptr, v16[t], N[t]);
            fill_f32_k<<<cdiv((long)N[t] * 64, 256), 256, 0, stream>>>(agg[t], 0.0f, N[t] * 64);
        }
        // relations: r=0 about (author->topic), r=1 rev_about (topic->author)
        for (int r = 0; r < 2; ++r) {
            const int src = (r == 0) ? 0 : 1;
            const int dst = 1 - src;
            const int* ei = (r == 0) ? eat : eta;
            gemm_wmma_f16<64, false, 0, false, true, false><<<cdiv(N[src], 64), 128, 0, stream>>>(
                k16[src], aRel16[l][r], nullptr, ktF, nullptr, N[src]);
            gemm_wmma_f16<64, false, 0, false, true, false><<<cdiv(N[src], 64), 128, 0, stream>>>(
                v16[src], mRel16[l][r], nullptr, vtF, nullptr, N[src]);
            fill_f32_k<<<cdiv(N[dst], 256), 256, 0, stream>>>(mbuf, -3.0e38f, N[dst]);
            fill_f32_k<<<cdiv(N[dst], 256), 256, 0, stream>>>(den, 0.0f, N[dst]);
            edge_alpha_k<<<cdiv(NE, 256), 256, 0, stream>>>(
                ei, NE, qF[dst], ktF, L[l].pRel[r], alpha, mbuf);
            edge_exp_k<<<cdiv(NE, 256), 256, 0, stream>>>(ei, NE, alpha, mbuf, den);
            edge_scatter_k<<<cdiv((long)NE * 32, 256), 256, 0, stream>>>(
                ei, NE, alpha, den, vtF, agg[dst]);
        }
        // per-type: gelu -> a-lin -> skip mix -> batchnorm (writes next x)
        for (int t = 0; t < 2; ++t) {
            float* aOut = (t == 0) ? ktF : vtF;   // reuse scratch
            gelu_cvt_k<<<cdiv((long)N[t] * 64, 256), 256, 0, stream>>>(
                agg[t], gelu16, N[t] * 64);
            gemm_wmma_f16<64, false, 0, true, true, false><<<cdiv(N[t], 64), 128, 0, stream>>>(
                gelu16, aW16[l][t], L[l].aB[t], aOut, nullptr, N[t]);
            skip_mix_k<<<cdiv((long)N[t] * 64, 256), 256, 0, stream>>>(
                aOut, xF[t], L[l].skip[t], agg[t], N[t] * 64);
            bn_stats_k<<<64, 256, 0, stream>>>(agg[t], N[t], mu, var);
            bn_apply_k<<<cdiv((long)N[t] * 64, 256), 256, 0, stream>>>(
                agg[t], mu, var, L[l].gamma, L[l].beta, xF[t], x16[t], N[t] * 64);
        }
    }

    // ---- decoder ---------------------------------------------------------
    decoder_k<<<cdiv(NEL, 64), 128, 0, stream>>>(
        xF[0], eli, decW116, decB1, decW2, decB2, (float*)d_out);
}
